// MultiHeadAttention_12790412607578
// MI455X (gfx1250) — compile-verified, matrix operand order checked by reading, not run on hardware
//
#include <hip/hip_runtime.h>
#include <hip/hip_bf16.h>

// ---------------------------------------------------------------------------
// MultiHeadAttention for MI455X (gfx1250, wave32, WMMA).
// All matrix math via v_wmma_f32_16x16x32_bf16 (f32 accumulate).
// Whole working set (~165MB) is L2-resident; compute-bound => bf16 WMMA.
// Attention: 4 waves x 2 q-tiles per block share each staged K/V chunk
// (128 queries per K/V read => ~8x less L2 traffic than 1 tile/wave).
// K staging uses GLOBAL_LOAD_ASYNC_TO_LDS_B128 (builtin exists; params are
// int4-vector pointers per the round-2 diagnostic).
// ---------------------------------------------------------------------------

typedef __bf16 bf16_t;
typedef __attribute__((ext_vector_type(16))) __bf16 v16bf;
typedef __attribute__((ext_vector_type(8)))  float  v8f;
typedef int v4i_vs __attribute__((vector_size(4 * sizeof(int))));

#define D_MODEL 1024
#define HEADS   16
#define DK      64
#define BATCH   4
#define SEQ     2048

#if defined(__HIP_DEVICE_COMPILE__) && \
    __has_builtin(__builtin_amdgcn_global_load_async_to_lds_b128)
#define HAVE_ASYNC_LDS 1
#define AS_GLOBAL __attribute__((address_space(1)))
#define AS_LOCAL  __attribute__((address_space(3)))
#else
#define HAVE_ASYNC_LDS 0
#endif

static __device__ inline v8f v8f_zero() {
  v8f z;
#pragma unroll
  for (int i = 0; i < 8; ++i) z[i] = 0.0f;
  return z;
}

// 16-bit A(16x32)/B(32x16) lane layout (ISA 7.12.2): lane half khalf=0/8,
// fragment elements 0..7 -> K = khalf+0..7, elements 8..15 -> K = 16+khalf+0..7.
// Both runs are 16B-contiguous => two b128 loads.
static __device__ inline v16bf load_frag(const bf16_t* __restrict__ rowptr, int khalf) {
  union { uint4 q[2]; v16bf v; } f;
  f.q[0] = *(const uint4*)(rowptr + khalf);
  f.q[1] = *(const uint4*)(rowptr + 16 + khalf);
  return f.v;
}

// fp32 source: vector loads + convert-to-bf16 in register.
static __device__ inline v16bf load_frag(const float* __restrict__ rowptr, int khalf) {
  float4 a = *(const float4*)(rowptr + khalf);
  float4 b = *(const float4*)(rowptr + khalf + 4);
  float4 c = *(const float4*)(rowptr + 16 + khalf);
  float4 d = *(const float4*)(rowptr + 16 + khalf + 4);
  v16bf f;
  f[0] = (bf16_t)a.x; f[1] = (bf16_t)a.y; f[2]  = (bf16_t)a.z; f[3]  = (bf16_t)a.w;
  f[4] = (bf16_t)b.x; f[5] = (bf16_t)b.y; f[6]  = (bf16_t)b.z; f[7]  = (bf16_t)b.w;
  f[8] = (bf16_t)c.x; f[9] = (bf16_t)c.y; f[10] = (bf16_t)c.z; f[11] = (bf16_t)c.w;
  f[12] = (bf16_t)d.x; f[13] = (bf16_t)d.y; f[14] = (bf16_t)d.z; f[15] = (bf16_t)d.w;
  return f;
}

// ---------------------------------------------------------------------------
// fp32 -> bf16 conversion (weights)
// ---------------------------------------------------------------------------
__global__ void cvt_f32_to_bf16(const float* __restrict__ src,
                                bf16_t* __restrict__ dst, int n) {
  int i = blockIdx.x * blockDim.x + threadIdx.x;
  if (i < n) dst[i] = (bf16_t)src[i];
}

// ---------------------------------------------------------------------------
// GEMM: Y[M,N] = X[M,K] @ W[N,K]^T. Wave: 16x64 strip, block: 4 waves = 64x64.
// ---------------------------------------------------------------------------
template <typename AT, bool SPLIT_HEADS>
__global__ __launch_bounds__(128)
void gemm_xwt(const AT* __restrict__ X, const bf16_t* __restrict__ W,
              bf16_t* __restrict__ Ybf, float* __restrict__ Yf, int K) {
  const int lane  = threadIdx.x & 31;
  const int wave  = threadIdx.x >> 5;
  const int m0    = blockIdx.y * 64 + wave * 16;
  const int n0    = blockIdx.x * 64;
  const int khalf = (lane >> 4) * 8;
  const int ln    = lane & 15;
  const int row   = m0 + ln;

  v8f acc[4];
#pragma unroll
  for (int t = 0; t < 4; ++t) acc[t] = v8f_zero();

  for (int k0 = 0; k0 < K; k0 += 32) {
    v16bf a = load_frag(X + (size_t)row * K + k0, khalf);
#pragma unroll
    for (int t = 0; t < 4; ++t) {
      v16bf b = load_frag(W + (size_t)(n0 + t * 16 + ln) * K + k0, khalf);
      acc[t] = __builtin_amdgcn_wmma_f32_16x16x32_bf16(
          false, a, false, b, (short)0, acc[t], false, false);
    }
  }

  const int mrow_off = (lane >> 4) * 8;
#pragma unroll
  for (int t = 0; t < 4; ++t) {
#pragma unroll
    for (int i = 0; i < 8; ++i) {
      const int m = m0 + i + mrow_off;
      const int n = n0 + t * 16 + ln;
      if (SPLIT_HEADS) {
        const int b  = m >> 11;          // m / SEQ
        const int s  = m & (SEQ - 1);
        const int h  = n >> 6;           // n / DK
        const int dk = n & (DK - 1);
        Ybf[(((size_t)(b * HEADS + h)) * SEQ + s) * DK + dk] = (bf16_t)acc[t][i];
      } else {
        Yf[(size_t)m * D_MODEL + n] = acc[t][i];
      }
    }
  }
}

// ---------------------------------------------------------------------------
// Flash attention. Block: 128 threads (4 waves) = 128 query rows of one head.
// Each wave: two 16-row q tiles. Shared K (row-major) + V (transposed) chunks.
// ---------------------------------------------------------------------------
__global__ __launch_bounds__(128)
void flash_attn(const bf16_t* __restrict__ Qh, const bf16_t* __restrict__ Kh,
                const bf16_t* __restrict__ Vh, bf16_t* __restrict__ Ctx) {
  const int tid  = threadIdx.x;
  const int lane = tid & 31;
  const int wave = tid >> 5;
  const int bh   = blockIdx.y;
  const int b    = bh / HEADS;
  const int h    = bh % HEADS;
  const int blk_q0 = blockIdx.x * 128;
  const int wq0    = blk_q0 + wave * 32;   // 32 query rows per wave (2 tiles)

  const bf16_t* Q  = Qh + (size_t)bh * SEQ * DK;
  const bf16_t* Kp = Kh + (size_t)bh * SEQ * DK;
  const bf16_t* Vp = Vh + (size_t)bh * SEQ * DK;

  // Ks: 32 keys x 64 dk, stride 72 (144B, 16B aligned, odd bank step)
  // Vt: transposed 64 dk x 32 keys, stride 48 (96B, 16B aligned)
  // Ps: per-wave 16 x 32 P tile, stride 48
  __shared__ __align__(16) bf16_t Ks[32 * 72];
  __shared__ __align__(16) bf16_t Vt[64 * 48];
  __shared__ __align__(16) bf16_t Ps[4][16 * 48];

  const int khalf    = (lane >> 4) * 8;
  const int mrow_off = khalf;           // C-layout row offset for this half
  const int ln       = lane & 15;

  v16bf aq[2][2];
#pragma unroll
  for (int qt = 0; qt < 2; ++qt) {
    const int qrow = wq0 + qt * 16 + ln;
    aq[qt][0] = load_frag(Q + (size_t)qrow * DK + 0,  khalf);
    aq[qt][1] = load_frag(Q + (size_t)qrow * DK + 32, khalf);
  }

  float m_r[2][8], l_r[2][8];
  v8f acc[2][4];
#pragma unroll
  for (int qt = 0; qt < 2; ++qt) {
#pragma unroll
    for (int i = 0; i < 8; ++i) { m_r[qt][i] = -3.0e38f; l_r[qt][i] = 0.0f; }
#pragma unroll
    for (int t = 0; t < 4; ++t) acc[qt][t] = v8f_zero();
  }

  const int blk_kmax = blk_q0 + 127;
  for (int j0 = 0; j0 <= blk_kmax; j0 += 32) {
    // ---- cooperative staging: 128 threads, 32B of K and V per thread ----
    {
      const int key = tid >> 2;  // 0..31
      const int cg  = tid & 3;   // 16-element group within the row
      const bf16_t* gk = Kp + (size_t)(j0 + key) * DK + cg * 16;
      bf16_t* lk = Ks + key * 72 + cg * 16;
#if HAVE_ASYNC_LDS
      __builtin_amdgcn_global_load_async_to_lds_b128(
          (AS_GLOBAL v4i_vs*)(uintptr_t)gk,
          (AS_LOCAL  v4i_vs*)(uint32_t)(uintptr_t)lk, 0, 0);
      __builtin_amdgcn_global_load_async_to_lds_b128(
          (AS_GLOBAL v4i_vs*)(uintptr_t)(gk + 8),
          (AS_LOCAL  v4i_vs*)(uint32_t)(uintptr_t)(lk + 8), 0, 0);
#else
      *(uint4*)lk       = *(const uint4*)gk;
      *(uint4*)(lk + 8) = *(const uint4*)(gk + 8);
#endif
      // V staged transposed (needs register round-trip anyway)
      union { uint4 q[2]; bf16_t e[16]; } vv;
      const bf16_t* gv = Vp + (size_t)(j0 + key) * DK + cg * 16;
      vv.q[0] = *(const uint4*)gv;
      vv.q[1] = *(const uint4*)(gv + 8);
#pragma unroll
      for (int e2 = 0; e2 < 16; ++e2) Vt[(cg * 16 + e2) * 48 + key] = vv.e[e2];
    }
#if HAVE_ASYNC_LDS
    asm volatile("s_wait_asynccnt 0x0" ::: "memory");
#endif
    __syncthreads();

#pragma unroll
    for (int qt = 0; qt < 2; ++qt) {
      const int q0t  = wq0 + qt * 16;
      const int kmax = q0t + 15;
      if (j0 <= kmax) {  // wave-uniform causal chunk skip
        // ---- scores: S = Q(16x64) @ K^T(64x16) per 16-key subtile ----
        v8f s0 = v8f_zero();
        v8f s1 = v8f_zero();
        {
          const bf16_t* krow = Ks + (size_t)ln * 72;
          v16bf bk0 = load_frag(krow + 0,  khalf);
          v16bf bk1 = load_frag(krow + 32, khalf);
          s0 = __builtin_amdgcn_wmma_f32_16x16x32_bf16(false, aq[qt][0], false, bk0,
                                                       (short)0, s0, false, false);
          s0 = __builtin_amdgcn_wmma_f32_16x16x32_bf16(false, aq[qt][1], false, bk1,
                                                       (short)0, s0, false, false);
        }
        const bool t1_active = (j0 + 16) <= kmax;
        if (t1_active) {
          const bf16_t* krow = Ks + (size_t)(16 + ln) * 72;
          v16bf bk0 = load_frag(krow + 0,  khalf);
          v16bf bk1 = load_frag(krow + 32, khalf);
          s1 = __builtin_amdgcn_wmma_f32_16x16x32_bf16(false, aq[qt][0], false, bk0,
                                                       (short)0, s1, false, false);
          s1 = __builtin_amdgcn_wmma_f32_16x16x32_bf16(false, aq[qt][1], false, bk1,
                                                       (short)0, s1, false, false);
        }

        // ---- causal mask + online softmax (row stats per 16-lane half) ----
        const float scale = 0.125f;  // 1/sqrt(64)
        float p0[8], p1[8], alpha[8];
#pragma unroll
        for (int i = 0; i < 8; ++i) {
          const int m    = q0t + i + mrow_off;
          const int key0 = j0 + ln;
          const int key1 = j0 + 16 + ln;
          const float v0 = (key0 <= m) ? s0[i] * scale : -3.0e38f;
          const float v1 = (t1_active && key1 <= m) ? s1[i] * scale : -3.0e38f;
          float rmax = fmaxf(v0, v1);
#pragma unroll
          for (int x = 8; x >= 1; x >>= 1)
            rmax = fmaxf(rmax, __shfl_xor(rmax, x, 32));
          const float mnew = fmaxf(m_r[qt][i], rmax);
          alpha[i] = __expf(m_r[qt][i] - mnew);
          p0[i]    = __expf(v0 - mnew);
          p1[i]    = __expf(v1 - mnew);
          float rsum = p0[i] + p1[i];
#pragma unroll
          for (int x = 8; x >= 1; x >>= 1) rsum += __shfl_xor(rsum, x, 32);
          l_r[qt][i] = l_r[qt][i] * alpha[i] + rsum;
          m_r[qt][i] = mnew;
        }

        // ---- spill P (C layout) -> LDS, re-read as A fragment.
        // Same-wave LDS ops are in-order: no barrier needed (per-wave buffer).
        bf16_t* Pw = Ps[wave];
#pragma unroll
        for (int i = 0; i < 8; ++i) {
          const int prow = i + mrow_off;
          Pw[prow * 48 + ln]      = (bf16_t)p0[i];
          Pw[prow * 48 + 16 + ln] = (bf16_t)(t1_active ? p1[i] : 0.0f);
        }

        // rescale running output by alpha before accumulating this chunk
#pragma unroll
        for (int t = 0; t < 4; ++t)
#pragma unroll
          for (int i = 0; i < 8; ++i) acc[qt][t][i] *= alpha[i];

        const v16bf pa = load_frag(Pw + (size_t)ln * 48, khalf);

        // ---- O += P(16x32) @ V(32x64); V fragments from transposed Vt ----
#pragma unroll
        for (int t = 0; t < 4; ++t) {
          v16bf vb = load_frag(Vt + (size_t)(t * 16 + ln) * 48, khalf);
          acc[qt][t] = __builtin_amdgcn_wmma_f32_16x16x32_bf16(
              false, pa, false, vb, (short)0, acc[qt][t], false, false);
        }
      }
    }
    __syncthreads();  // all waves done reading Ks/Vt before restaging
  }

  // ---- normalize and store ctx in [B,S,D_MODEL] bf16 layout ----
#pragma unroll
  for (int qt = 0; qt < 2; ++qt) {
#pragma unroll
    for (int i = 0; i < 8; ++i) {
      const float inv = 1.0f / l_r[qt][i];
      const int s = wq0 + qt * 16 + i + mrow_off;
#pragma unroll
      for (int t = 0; t < 4; ++t) {
        const int dk = t * 16 + ln;
        Ctx[((size_t)(b * SEQ + s)) * D_MODEL + h * DK + dk] =
            (bf16_t)(acc[qt][t][i] * inv);
      }
    }
  }
}

// ---------------------------------------------------------------------------
// Host launcher. Workspace layout (72 MB):
//   4 x bf16 weights (2MB) | qh | kh | vh (16MB each) | ctx (16MB)
// ---------------------------------------------------------------------------
extern "C" void kernel_launch(void* const* d_in, const int* in_sizes, int n_in,
                              void* d_out, int out_size, void* d_ws, size_t ws_size,
                              hipStream_t stream) {
  const float* q  = (const float*)d_in[0];
  const float* k  = (const float*)d_in[1];
  const float* v  = (const float*)d_in[2];
  // d_in[3] is the boolean causal mask; masking is done analytically.
  const float* Wq = (const float*)d_in[4];
  const float* Wk = (const float*)d_in[5];
  const float* Wv = (const float*)d_in[6];
  const float* Wo = (const float*)d_in[7];

  const size_t NW = (size_t)D_MODEL * D_MODEL;
  const size_t NH = (size_t)BATCH * HEADS * SEQ * DK;
  const size_t NC = (size_t)BATCH * SEQ * D_MODEL;

  char* ws = (char*)d_ws;
  size_t off = 0;
  bf16_t* Wq_b = (bf16_t*)(ws + off); off += NW * sizeof(bf16_t);
  bf16_t* Wk_b = (bf16_t*)(ws + off); off += NW * sizeof(bf16_t);
  bf16_t* Wv_b = (bf16_t*)(ws + off); off += NW * sizeof(bf16_t);
  bf16_t* Wo_b = (bf16_t*)(ws + off); off += NW * sizeof(bf16_t);
  bf16_t* qh   = (bf16_t*)(ws + off); off += NH * sizeof(bf16_t);
  bf16_t* kh   = (bf16_t*)(ws + off); off += NH * sizeof(bf16_t);
  bf16_t* vh   = (bf16_t*)(ws + off); off += NH * sizeof(bf16_t);
  bf16_t* ctx  = (bf16_t*)(ws + off); off += NC * sizeof(bf16_t);

  {
    const int n = (int)NW;
    const int blocks = (n + 255) / 256;
    cvt_f32_to_bf16<<<blocks, 256, 0, stream>>>(Wq, Wq_b, n);
    cvt_f32_to_bf16<<<blocks, 256, 0, stream>>>(Wk, Wk_b, n);
    cvt_f32_to_bf16<<<blocks, 256, 0, stream>>>(Wv, Wv_b, n);
    cvt_f32_to_bf16<<<blocks, 256, 0, stream>>>(Wo, Wo_b, n);
  }

  {
    dim3 grid(D_MODEL / 64, (BATCH * SEQ) / 64);
    gemm_xwt<float, true><<<grid, 128, 0, stream>>>(q, Wq_b, qh, nullptr, D_MODEL);
    gemm_xwt<float, true><<<grid, 128, 0, stream>>>(k, Wk_b, kh, nullptr, D_MODEL);
    gemm_xwt<float, true><<<grid, 128, 0, stream>>>(v, Wv_b, vh, nullptr, D_MODEL);
  }

  {
    dim3 grid(SEQ / 128, BATCH * HEADS);
    flash_attn<<<grid, 128, 0, stream>>>(qh, kh, vh, ctx);
  }

  {
    dim3 grid(D_MODEL / 64, (BATCH * SEQ) / 64);
    gemm_xwt<bf16_t, false><<<grid, 128, 0, stream>>>(ctx, Wo_b, nullptr,
                                                      (float*)d_out, D_MODEL);
  }
}
